// TransformerNoTokenGenerator_43731357008454
// MI455X (gfx1250) — matrix operand order repaired and checked by reading, not compiled, not run on hardware
//
#include <hip/hip_runtime.h>
#include <hip/hip_bf16.h>
#include <math.h>

// ---------------- problem constants ----------------
#define NLAYERS 6
#define DMODEL  512
#define NHEAD   8
#define HEADDIM 64
#define DFF     2048
#define BATCH   8
#define SEQ     1024
#define ROWS    (BATCH*SEQ)      // 8192 token rows
#define QKVDIM  (3*DMODEL)       // 1536

// async global->LDS path (gfx1250) if the toolchain exposes the builtins
#if defined(__has_builtin)
#if __has_builtin(__builtin_amdgcn_global_load_async_to_lds_b128) && \
    __has_builtin(__builtin_amdgcn_s_wait_asynccnt)
#define USE_ASYNC_LDS 1
#endif
#endif
#ifdef USE_ASYNC_LDS
#define AS1 __attribute__((address_space(1)))
#define AS3 __attribute__((address_space(3)))
typedef __attribute__((__vector_size__(4 * sizeof(int)))) int v4i_t;
// async copy of 16B: src in global (AS1), dst in LDS (AS3)
#define ASYNC_CP16(dst, src) \
  __builtin_amdgcn_global_load_async_to_lds_b128((AS1 v4i_t*)(src), (AS3 v4i_t*)(dst), 0, 0)
#endif

// ---------------- types for WMMA ----------------
typedef __attribute__((ext_vector_type(16))) __bf16 v16bf_t;
typedef __attribute__((ext_vector_type(8)))  float  v8f_t;

union FragB {
  v16bf_t v;
  uint4   q[2];
  unsigned short h[16];
};

__device__ __forceinline__ unsigned short f2bf(float f) {
  union { float f; unsigned u; } x; x.f = f;
  unsigned r = x.u + 0x7FFFu + ((x.u >> 16) & 1u);   // RNE
  return (unsigned short)(r >> 16);
}
__device__ __forceinline__ float bf2f(unsigned short h) {
  union { unsigned u; float f; } x; x.u = ((unsigned)h) << 16;
  return x.f;
}
__device__ __forceinline__ v8f_t wmma_bf16(v16bf_t a, v16bf_t b, v8f_t c) {
  // D = A(16x32 bf16) * B(32x16 bf16) + C(16x16 f32)
  return __builtin_amdgcn_wmma_f32_16x16x32_bf16(false, a, false, b, (short)0, c,
                                                 false, false);
}
// exact-GELU via branch-free erf approximation (A&S 7.1.26, |err|<1.5e-7)
__device__ __forceinline__ float gelu_f(float x) {
  const float ax = fabsf(x) * 0.70710678118654752f;
  const float t  = __builtin_amdgcn_rcpf(1.f + 0.3275911f * ax);
  const float poly = t * (0.254829592f + t * (-0.284496736f + t * (1.421413741f +
                     t * (-1.453152027f + t * 1.061405429f))));
  const float erfax = 1.f - poly * __expf(-ax * ax);
  return 0.5f * x * (1.f + copysignf(erfax, x));
}

// ---------------- fp32 -> bf16 conversion ----------------
__global__ void cvt_kernel(const float* __restrict__ in,
                           unsigned short* __restrict__ out, long n4) {
  long i = (long)blockIdx.x * blockDim.x + threadIdx.x;
  if (i >= n4) return;
  float4 v = reinterpret_cast<const float4*>(in)[i];
  uint2 o;
  o.x = (unsigned)f2bf(v.x) | ((unsigned)f2bf(v.y) << 16);
  o.y = (unsigned)f2bf(v.z) | ((unsigned)f2bf(v.w) << 16);
  reinterpret_cast<uint2*>(out)[i] = o;
}

// ---------------- key-padding mask: kpm[row]=1 iff x row all-zero ----------------
__global__ __launch_bounds__(128)
void kpm_kernel(const float* __restrict__ x, int* __restrict__ kpm) {
  __shared__ int flag;
  if (threadIdx.x == 0) flag = 0;
  __syncthreads();
  const float4* xr = reinterpret_cast<const float4*>(x + (size_t)blockIdx.x * DMODEL);
  float4 v = xr[threadIdx.x];
  if (v.x != 0.f || v.y != 0.f || v.z != 0.f || v.w != 0.f) atomicOr(&flag, 1);
  __syncthreads();
  if (threadIdx.x == 0) kpm[blockIdx.x] = (flag == 0) ? 1 : 0;
}

// ---------------- V transpose: qkv(B,L,1536) v-section -> vT(B,H,HD,L) ----------------
__global__ __launch_bounds__(256)
void vtrans_kernel(const unsigned short* __restrict__ qkv,
                   unsigned short* __restrict__ vT) {
  const int row  = blockIdx.x;            // b*SEQ + kpos
  const int b    = row >> 10;
  const int kpos = row & (SEQ - 1);
  const int c    = threadIdx.x * 2;       // 0..510, even -> same head for both halves
  unsigned u = *reinterpret_cast<const unsigned*>(
      qkv + (size_t)row * QKVDIM + 2 * DMODEL + c);
  const int h0 = c >> 6, d0 = c & 63;
  const size_t base = (size_t)(b * NHEAD + h0) * HEADDIM;
  vT[(base + d0)     * SEQ + kpos] = (unsigned short)(u & 0xFFFFu);
  vT[(base + d0 + 1) * SEQ + kpos] = (unsigned short)(u >> 16);
}

// ---------------- generic bf16 WMMA GEMM: C[M,N] = A[M,K] * W[N,K]^T + bias ----------------
// Block tile 128x64, 4 waves; W tile double-buffered in LDS via async global->LDS;
// K-loop unrolled 2x with fixed stage roles (no register rotation); DS loads batched
// ahead of each 8-WMMA burst. Requires K % 64 == 0 (512 / 2048 here).
// mode 0: store bf16   mode 1: GELU then bf16   mode 2: store f32
__global__ __launch_bounds__(128)
void gemm_bf16_kernel(const unsigned short* __restrict__ A,
                      const unsigned short* __restrict__ W,
                      const float* __restrict__ bias,
                      float* __restrict__ Cf,
                      unsigned short* __restrict__ Cb,
                      int M, int N, int K, int mode) {
  __shared__ __align__(16) unsigned short Wl[2][64][40];  // 64 n-rows x 32 k-halfs (+pad)
  const int n0   = blockIdx.x * 64;
  const int m0   = blockIdx.y * 128;
  const int tid  = threadIdx.x;
  const int wv   = tid >> 5;
  const int lane = tid & 31;
  const int g    = lane >> 4;      // half-wave group
  const int r    = lane & 15;      // row (A) / column (B,C)

  // cooperative W staging: thread t covers 16 halfs of row (t>>1), chunk (t&1)*16
  const int wn = tid >> 1;
  const int wc = (tid & 1) * 16;
  const unsigned short* Wrow = W + (size_t)(n0 + wn) * K + wc;
  const unsigned short* Arow0 = A + (size_t)(m0 + wv * 32 + r)      * K + g * 8;
  const unsigned short* Arow1 = A + (size_t)(m0 + wv * 32 + 16 + r) * K + g * 8;

  v8f_t acc[2][4];
#pragma unroll
  for (int i2 = 0; i2 < 2; ++i2)
#pragma unroll
    for (int j = 0; j < 4; ++j) acc[i2][j] = (v8f_t){0,0,0,0,0,0,0,0};

  // one K-step: batched LDS loads from stage st, then dense 8-WMMA burst
  auto mmstep = [&](int st, const FragB* aa) {
    FragB bfr[4];
#pragma unroll
    for (int j = 0; j < 4; ++j) {
      bfr[j].q[0] = *reinterpret_cast<const uint4*>(&Wl[st][j * 16 + r][g * 8]);
      bfr[j].q[1] = *reinterpret_cast<const uint4*>(&Wl[st][j * 16 + r][g * 8 + 16]);
    }
#pragma unroll
    for (int j = 0; j < 4; ++j) acc[0][j] = wmma_bf16(aa[0].v, bfr[j].v, acc[0][j]);
#pragma unroll
    for (int j = 0; j < 4; ++j) acc[1][j] = wmma_bf16(aa[1].v, bfr[j].v, acc[1][j]);
  };

  // ---- prologue: stage 0 <- k=0; A fragments for k=0 ----
#ifdef USE_ASYNC_LDS
  ASYNC_CP16(&Wl[0][wn][wc],     Wrow);
  ASYNC_CP16(&Wl[0][wn][wc + 8], Wrow + 8);
  __builtin_amdgcn_s_wait_asynccnt(0);
#else
  {
    uint4 w0 = *reinterpret_cast<const uint4*>(Wrow);
    uint4 w1 = *reinterpret_cast<const uint4*>(Wrow + 8);
    *reinterpret_cast<uint4*>(&Wl[0][wn][wc])     = w0;
    *reinterpret_cast<uint4*>(&Wl[0][wn][wc + 8]) = w1;
  }
#endif
  FragB a[2], an[2];
  a[0].q[0] = *reinterpret_cast<const uint4*>(Arow0);
  a[0].q[1] = *reinterpret_cast<const uint4*>(Arow0 + 16);
  a[1].q[0] = *reinterpret_cast<const uint4*>(Arow1);
  a[1].q[1] = *reinterpret_cast<const uint4*>(Arow1 + 16);
  __syncthreads();

  for (int k0 = 0; k0 < K; k0 += 64) {
    // ---- step A: compute k0 on stage 0; prefetch k0+32 -> stage 1 + an ----
#ifdef USE_ASYNC_LDS
    ASYNC_CP16(&Wl[1][wn][wc],     Wrow + k0 + 32);
    ASYNC_CP16(&Wl[1][wn][wc + 8], Wrow + k0 + 40);
#else
    uint4 nw0 = *reinterpret_cast<const uint4*>(Wrow + k0 + 32);
    uint4 nw1 = *reinterpret_cast<const uint4*>(Wrow + k0 + 40);
#endif
    __builtin_prefetch(Arow0 + k0 + 192, 0, 1);      // global_prefetch_b8
    an[0].q[0] = *reinterpret_cast<const uint4*>(Arow0 + k0 + 32);
    an[0].q[1] = *reinterpret_cast<const uint4*>(Arow0 + k0 + 48);
    an[1].q[0] = *reinterpret_cast<const uint4*>(Arow1 + k0 + 32);
    an[1].q[1] = *reinterpret_cast<const uint4*>(Arow1 + k0 + 48);
    mmstep(0, a);
#ifdef USE_ASYNC_LDS
    __builtin_amdgcn_s_wait_asynccnt(0);
#else
    *reinterpret_cast<uint4*>(&Wl[1][wn][wc])     = nw0;
    *reinterpret_cast<uint4*>(&Wl[1][wn][wc + 8]) = nw1;
#endif
    __syncthreads();

    // ---- step B: compute k0+32 on stage 1; prefetch k0+64 -> stage 0 + a ----
    const bool more = (k0 + 64) < K;
#ifndef USE_ASYNC_LDS
    uint4 mw0, mw1;
#endif
    if (more) {
#ifdef USE_ASYNC_LDS
      ASYNC_CP16(&Wl[0][wn][wc],     Wrow + k0 + 64);
      ASYNC_CP16(&Wl[0][wn][wc + 8], Wrow + k0 + 72);
#else
      mw0 = *reinterpret_cast<const uint4*>(Wrow + k0 + 64);
      mw1 = *reinterpret_cast<const uint4*>(Wrow + k0 + 72);
#endif
      a[0].q[0] = *reinterpret_cast<const uint4*>(Arow0 + k0 + 64);
      a[0].q[1] = *reinterpret_cast<const uint4*>(Arow0 + k0 + 80);
      a[1].q[0] = *reinterpret_cast<const uint4*>(Arow1 + k0 + 64);
      a[1].q[1] = *reinterpret_cast<const uint4*>(Arow1 + k0 + 80);
    }
    mmstep(1, an);
    if (more) {
#ifdef USE_ASYNC_LDS
      __builtin_amdgcn_s_wait_asynccnt(0);
#else
      *reinterpret_cast<uint4*>(&Wl[0][wn][wc])     = mw0;
      *reinterpret_cast<uint4*>(&Wl[0][wn][wc + 8]) = mw1;
#endif
      __syncthreads();
    }
  }
  // epilogue: C/D layout -> row = v + 8*g, col = r
#pragma unroll
  for (int i2 = 0; i2 < 2; ++i2) {
#pragma unroll
    for (int j = 0; j < 4; ++j) {
      const int n = n0 + j * 16 + r;
      const float bv = bias[n];
#pragma unroll
      for (int v = 0; v < 8; ++v) {
        const int m = m0 + wv * 32 + i2 * 16 + v + g * 8;
        float val = acc[i2][j][v] + bv;
        if (mode == 1) val = gelu_f(val);
        if (mode == 2) Cf[(size_t)m * N + n] = val;
        else           Cb[(size_t)m * N + n] = f2bf(val);
      }
    }
  }
}

// ---------------- attention: per (b, h, 16-row q tile); causal + rel-pos bias + kpm ----------------
__global__ __launch_bounds__(128)
void attn_kernel(const unsigned short* __restrict__ qkv,   // (B,L,1536) bf16
                 const unsigned short* __restrict__ vT,    // (B,H,HD,L) bf16
                 const float* __restrict__ dist_emb,       // (1025,8) f32
                 const int* __restrict__ kpm,              // (B*L)
                 unsigned short* __restrict__ attnb) {     // (B,L,512) bf16
  __shared__ __align__(16) unsigned short Sp[16][1032];    // score/prob strip (bf16)
  __shared__ float rsum[16];

  const int qt = blockIdx.x, h = blockIdx.y, b = blockIdx.z;
  const int q0 = qt * 16;
  const int ntiles = qt + 1;                 // causal: k-tiles up to diagonal
  const int kmax  = ntiles * 16;
  const int kmax2 = (kmax + 31) & ~31;       // round up to WMMA K step
  const int wv   = threadIdx.x >> 5;
  const int lane = threadIdx.x & 31;
  const int g    = lane >> 4;
  const int r    = lane & 15;
  const size_t rstride = QKVDIM;
  const float scale = 0.125f;                // 1/sqrt(64)

  // Q fragments for d=[0,32) and [32,64)
  const unsigned short* Qrow = qkv + (size_t)(b * SEQ + q0 + r) * rstride + h * HEADDIM;
  FragB qa0, qa1;
  qa0.q[0] = *reinterpret_cast<const uint4*>(Qrow + g * 8);
  qa0.q[1] = *reinterpret_cast<const uint4*>(Qrow + g * 8 + 16);
  qa1.q[0] = *reinterpret_cast<const uint4*>(Qrow + 32 + g * 8);
  qa1.q[1] = *reinterpret_cast<const uint4*>(Qrow + 32 + g * 8 + 16);

  // ---- phase 1: S = scale*Q.K^T + bias (masked); pipelined over k-tiles ----
  {
    int t = wv;
    FragB kb0, kb1;
    if (t < ntiles) {
      const unsigned short* Krow =
          qkv + (size_t)(b * SEQ + t * 16 + r) * rstride + DMODEL + h * HEADDIM;
      kb0.q[0] = *reinterpret_cast<const uint4*>(Krow + g * 8);
      kb0.q[1] = *reinterpret_cast<const uint4*>(Krow + g * 8 + 16);
      kb1.q[0] = *reinterpret_cast<const uint4*>(Krow + 32 + g * 8);
      kb1.q[1] = *reinterpret_cast<const uint4*>(Krow + 32 + g * 8 + 16);
    }
    for (; t < ntiles; t += 4) {
      const int tn = t + 4;
      FragB nb0, nb1;
      if (tn < ntiles) {
        const unsigned short* Krow =
            qkv + (size_t)(b * SEQ + tn * 16 + r) * rstride + DMODEL + h * HEADDIM;
        nb0.q[0] = *reinterpret_cast<const uint4*>(Krow + g * 8);
        nb0.q[1] = *reinterpret_cast<const uint4*>(Krow + g * 8 + 16);
        nb1.q[0] = *reinterpret_cast<const uint4*>(Krow + 32 + g * 8);
        nb1.q[1] = *reinterpret_cast<const uint4*>(Krow + 32 + g * 8 + 16);
      }
      v8f_t acc = (v8f_t){0,0,0,0,0,0,0,0};
      acc = wmma_bf16(qa0.v, kb0.v, acc);
      acc = wmma_bf16(qa1.v, kb1.v, acc);
      const int k0 = t * 16;
#pragma unroll
      for (int v = 0; v < 8; ++v) {
        const int q = q0 + v + g * 8;
        const int k = k0 + r;
        float sv;
        if (k > q || kpm[b * SEQ + k]) sv = -__builtin_inff();
        else                           sv = acc[v] * scale + dist_emb[(q - k) * NHEAD + h];
        Sp[v + g * 8][k] = f2bf(sv);
      }
      if (tn < ntiles) { kb0 = nb0; kb1 = nb1; }
    }
  }
  __syncthreads();

  // ---- phase 2: row softmax (8 threads per row) ----
  {
    const int row = threadIdx.x >> 3;
    const int j   = threadIdx.x & 7;
    float mx = -__builtin_inff();
    for (int c = j; c < kmax; c += 8) mx = fmaxf(mx, bf2f(Sp[row][c]));
    for (int off = 4; off; off >>= 1) mx = fmaxf(mx, __shfl_down(mx, off, 8));
    mx = __shfl(mx, 0, 8);
    float sm = 0.f;
    for (int c = j; c < kmax; c += 8) {
      float pv = __expf(bf2f(Sp[row][c]) - mx);
      sm += pv;
      Sp[row][c] = f2bf(pv);
    }
    for (int c = kmax + j; c < kmax2; c += 8) Sp[row][c] = 0;  // zero WMMA tail
    for (int off = 4; off; off >>= 1) sm += __shfl_down(sm, off, 8);
    if (j == 0) rsum[row] = sm;
  }
  __syncthreads();

  // ---- phase 3: out = (P.V)/rsum; wave wv owns d-subtile; P and V both pipelined ----
  {
    const int n0d = wv * 16;
    const unsigned short* Vrow =
        vT + ((size_t)(b * NHEAD + h) * HEADDIM + n0d + r) * SEQ + g * 8;
    v8f_t acc = (v8f_t){0,0,0,0,0,0,0,0};
    FragB vb, pa;
    vb.q[0] = *reinterpret_cast<const uint4*>(Vrow);
    vb.q[1] = *reinterpret_cast<const uint4*>(Vrow + 16);
    pa.q[0] = *reinterpret_cast<const uint4*>(&Sp[r][g * 8]);       // ds_load_b128
    pa.q[1] = *reinterpret_cast<const uint4*>(&Sp[r][g * 8 + 16]);
    for (int kk = 0; kk < kmax2; kk += 32) {
      FragB nvb, npa;
      const bool more = (kk + 32) < kmax2;
      if (more) {
        nvb.q[0] = *reinterpret_cast<const uint4*>(Vrow + kk + 32);
        nvb.q[1] = *reinterpret_cast<const uint4*>(Vrow + kk + 48);
        npa.q[0] = *reinterpret_cast<const uint4*>(&Sp[r][kk + 32 + g * 8]);
        npa.q[1] = *reinterpret_cast<const uint4*>(&Sp[r][kk + 32 + g * 8 + 16]);
      }
      acc = wmma_bf16(pa.v, vb.v, acc);
      if (more) { vb = nvb; pa = npa; }
    }
#pragma unroll
    for (int v = 0; v < 8; ++v) {
      const int q = q0 + v + g * 8;
      const float val = acc[v] / rsum[v + g * 8];
      attnb[(size_t)(b * SEQ + q) * DMODEL + h * HEADDIM + n0d + r] = f2bf(val);
    }
  }
}

// ---------------- residual + LayerNorm (fp32), one row per block ----------------
__global__ __launch_bounds__(128)
void ln_kernel(const float* __restrict__ res,   // may be null
               const float* __restrict__ y,
               const float* __restrict__ gam,
               const float* __restrict__ bet,
               float* __restrict__ outp) {
  __shared__ float sh[8];
  const size_t row = blockIdx.x;
  const int tid = threadIdx.x;
  float4 v = *reinterpret_cast<const float4*>(y + row * DMODEL + tid * 4);
  if (res) {
    float4 a = *reinterpret_cast<const float4*>(res + row * DMODEL + tid * 4);
    v.x += a.x; v.y += a.y; v.z += a.z; v.w += a.w;
  }
  float s  = v.x + v.y + v.z + v.w;
  float s2 = v.x*v.x + v.y*v.y + v.z*v.z + v.w*v.w;
  for (int off = 16; off; off >>= 1) { s += __shfl_down(s, off, 32); s2 += __shfl_down(s2, off, 32); }
  if ((tid & 31) == 0) { sh[tid >> 5] = s; sh[4 + (tid >> 5)] = s2; }
  __syncthreads();
  s  = sh[0] + sh[1] + sh[2] + sh[3];
  s2 = sh[4] + sh[5] + sh[6] + sh[7];
  const float mean = s * (1.f / DMODEL);
  const float var  = s2 * (1.f / DMODEL) - mean * mean;
  const float inv  = rsqrtf(var + 1e-5f);
  float4 gv = *reinterpret_cast<const float4*>(gam + tid * 4);
  float4 bv = *reinterpret_cast<const float4*>(bet + tid * 4);
  float4 o;
  o.x = (v.x - mean) * inv * gv.x + bv.x;
  o.y = (v.y - mean) * inv * gv.y + bv.y;
  o.z = (v.z - mean) * inv * gv.z + bv.z;
  o.w = (v.w - mean) * inv * gv.w + bv.w;
  *reinterpret_cast<float4*>(outp + row * DMODEL + tid * 4) = o;
}

// ---------------- host side ----------------
extern "C" void kernel_launch(void* const* d_in, const int* in_sizes, int n_in,
                              void* d_out, int out_size, void* d_ws, size_t ws_size,
                              hipStream_t stream) {
  (void)in_sizes; (void)n_in; (void)out_size; (void)ws_size;
  const float* x         = (const float*)d_in[0];
  const float* dist_emb  = (const float*)d_in[1];
  const float* in_proj_w = (const float*)d_in[2];
  const float* in_proj_b = (const float*)d_in[3];
  const float* out_w     = (const float*)d_in[4];
  const float* out_b     = (const float*)d_in[5];
  const float* lin1_w    = (const float*)d_in[6];
  const float* lin1_b    = (const float*)d_in[7];
  const float* lin2_w    = (const float*)d_in[8];
  const float* lin2_b    = (const float*)d_in[9];
  const float* ln1_g     = (const float*)d_in[10];
  const float* ln1_bb    = (const float*)d_in[11];
  const float* ln2_g     = (const float*)d_in[12];
  const float* ln2_bb    = (const float*)d_in[13];
  const float* lnf_g     = (const float*)d_in[14];
  const float* lnf_bb    = (const float*)d_in[15];

  char* p = (char*)d_ws;
  auto take = [&](size_t bytes) -> char* {
    char* r = p; p += (bytes + 255) & ~(size_t)255; return r;
  };
  const size_t W_IN  = (size_t)NLAYERS * QKVDIM * DMODEL;
  const size_t W_OUT = (size_t)NLAYERS * DMODEL * DMODEL;
  const size_t W_L1  = (size_t)NLAYERS * DFF    * DMODEL;
  const size_t W_L2  = (size_t)NLAYERS * DMODEL * DFF;
  unsigned short* inWb  = (unsigned short*)take(W_IN  * 2);
  unsigned short* outWb = (unsigned short*)take(W_OUT * 2);
  unsigned short* l1Wb  = (unsigned short*)take(W_L1  * 2);
  unsigned short* l2Wb  = (unsigned short*)take(W_L2  * 2);
  int*   kpm   = (int*)  take((size_t)ROWS * 4);
  float* cur   = (float*)take((size_t)ROWS * DMODEL * 4);
  unsigned short* curb  = (unsigned short*)take((size_t)ROWS * DMODEL * 2);
  unsigned short* qkvb  = (unsigned short*)take((size_t)ROWS * QKVDIM * 2);
  unsigned short* vTb   = (unsigned short*)take((size_t)ROWS * DMODEL * 2);
  unsigned short* attnb = (unsigned short*)take((size_t)ROWS * DMODEL * 2);
  float* tmp   = (float*)take((size_t)ROWS * DMODEL * 4);
  unsigned short* hb    = (unsigned short*)take((size_t)ROWS * DFF * 2);

  auto cvt = [&](const float* src, unsigned short* dst, size_t n) {
    size_t n4 = n / 4;
    cvt_kernel<<<dim3((unsigned)((n4 + 255) / 256)), dim3(256), 0, stream>>>(src, dst, (long)n4);
  };

  // one-time per launch: bf16 weights, key-padding mask, activation init
  cvt(in_proj_w, inWb,  W_IN);
  cvt(out_w,     outWb, W_OUT);
  cvt(lin1_w,    l1Wb,  W_L1);
  cvt(lin2_w,    l2Wb,  W_L2);
  kpm_kernel<<<dim3(ROWS), dim3(128), 0, stream>>>(x, kpm);
  (void)hipMemcpyAsync(cur, x, (size_t)ROWS * DMODEL * 4,
                       hipMemcpyDeviceToDevice, stream);

  for (int l = 0; l < NLAYERS; ++l) {
    // QKV projection -> bf16
    cvt(cur, curb, (size_t)ROWS * DMODEL);
    gemm_bf16_kernel<<<dim3(QKVDIM / 64, ROWS / 128), dim3(128), 0, stream>>>(
        curb, inWb + (size_t)l * QKVDIM * DMODEL, in_proj_b + (size_t)l * QKVDIM,
        nullptr, qkvb, ROWS, QKVDIM, DMODEL, 0);
    // V transpose for vectorized PV loads
    vtrans_kernel<<<dim3(ROWS), dim3(256), 0, stream>>>(qkvb, vTb);
    // attention
    attn_kernel<<<dim3(SEQ / 16, NHEAD, BATCH), dim3(128), 0, stream>>>(
        qkvb, vTb, dist_emb, kpm, attnb);
    // out projection -> f32, then residual+LN1
    gemm_bf16_kernel<<<dim3(DMODEL / 64, ROWS / 128), dim3(128), 0, stream>>>(
        attnb, outWb + (size_t)l * DMODEL * DMODEL, out_b + (size_t)l * DMODEL,
        tmp, nullptr, ROWS, DMODEL, DMODEL, 2);
    ln_kernel<<<dim3(ROWS), dim3(128), 0, stream>>>(
        cur, tmp, ln1_g + (size_t)l * DMODEL, ln1_bb + (size_t)l * DMODEL, cur);
    // FFN: lin1 + GELU -> bf16, lin2 -> f32, residual+LN2
    cvt(cur, curb, (size_t)ROWS * DMODEL);
    gemm_bf16_kernel<<<dim3(DFF / 64, ROWS / 128), dim3(128), 0, stream>>>(
        curb, l1Wb + (size_t)l * DFF * DMODEL, lin1_b + (size_t)l * DFF,
        nullptr, hb, ROWS, DFF, DMODEL, 1);
    gemm_bf16_kernel<<<dim3(DMODEL / 64, ROWS / 128), dim3(128), 0, stream>>>(
        hb, l2Wb + (size_t)l * DMODEL * DFF, lin2_b + (size_t)l * DMODEL,
        tmp, nullptr, ROWS, DMODEL, DFF, 2);
    ln_kernel<<<dim3(ROWS), dim3(128), 0, stream>>>(
        cur, tmp, ln2_g + (size_t)l * DMODEL, ln2_bb + (size_t)l * DMODEL, cur);
  }
  // final LN -> d_out
  ln_kernel<<<dim3(ROWS), dim3(128), 0, stream>>>(
      nullptr, cur, lnf_g, lnf_bb, (float*)d_out);
}
